// LocalSLC_48438641164552
// MI455X (gfx1250) — matrix operand order repaired and checked by reading, not compiled, not run on hardware
//
#include <hip/hip_runtime.h>

#define Bn 16
#define Nn 10000
#define Kk 16
#define CIN 64
#define COUT 64
#define ASTRIDE 72   // halves; 144B row stride: 16B-aligned b128 loads, bank-conflict-free

typedef __attribute__((ext_vector_type(16))) _Float16 v16h;
typedef __attribute__((ext_vector_type(8)))  _Float16 v8h;
typedef __attribute__((ext_vector_type(4)))  _Float16 v4h;
typedef __attribute__((ext_vector_type(8)))  float    v8f;

// One-time (per launch) conversion: ws[k][i][o] f32 -> Wt[k][o][i] f16 (B-operand friendly)
__global__ __launch_bounds__(256) void prep_wt(const float* __restrict__ ws,
                                               _Float16* __restrict__ wt) {
  int idx = blockIdx.x * 256 + threadIdx.x;
  if (idx >= Kk * CIN * COUT) return;
  int k   = idx >> 12;
  int rem = idx & 4095;
  int i   = rem >> 6;
  int o   = rem & 63;
  wt[(k << 12) + (o << 6) + i] = (_Float16)ws[idx];
}

// One workgroup per point n: 4 waves, wave w computes the 16x16 output tile for
// o in [16w, 16w+16). Y_n = sum_k bs[n,k] * Xg_k(16x64) @ W_k(64x64), bs folded into A.
__global__ __launch_bounds__(128) void slc_kernel(const float* __restrict__ x,
                                                  const int* __restrict__ knn,
                                                  const float* __restrict__ bsv,
                                                  const _Float16* __restrict__ wt,
                                                  float* __restrict__ y) {
  __shared__ __align__(16) _Float16 sA[Kk * Bn * ASTRIDE]; // ~36.9 KB staged f16 A tiles
  __shared__ int   s_id[Kk];
  __shared__ float s_bs[Kk];

  const int n   = blockIdx.x;
  const int tid = threadIdx.x;

  if (tid < Kk) {
    s_id[tid] = knn[n * Kk + tid];
    s_bs[tid] = bsv[n * Kk + tid];
  }
  __syncthreads();

  // Gather + scale + f32->f16 convert: 4096 float4 chunks (k,b,ic), 32 per thread.
  // Consecutive tids read consecutive 16B chunks of a gathered row -> coalesced.
#pragma unroll 4
  for (int it = 0; it < 32; ++it) {
    int c  = it * 128 + tid;      // 0..4095
    int k  = c >> 8;
    int b  = (c >> 4) & 15;
    int ic = c & 15;
    int   g = s_id[k];
    float s = s_bs[k];
    const float4 v = *(const float4*)(x + (((size_t)b * Nn + g) << 6) + (ic << 2));
    v4h hv;
    hv[0] = (_Float16)(v.x * s);
    hv[1] = (_Float16)(v.y * s);
    hv[2] = (_Float16)(v.z * s);
    hv[3] = (_Float16)(v.w * s);
    *(v4h*)(&sA[(k * Bn + b) * ASTRIDE + (ic << 2)]) = hv;
  }
  __syncthreads();

  const int wv   = tid >> 5;    // wave -> o tile
  const int lane = tid & 31;
  const int l16  = lane & 15;
  const int lhi  = lane >> 4;

  v8f acc = {};
#pragma unroll 4
  for (int k = 0; k < Kk; ++k) {
#pragma unroll
    for (int h = 0; h < 2; ++h) {    // two 16x16x32 K-steps cover C_IN=64
      // A frag (ISA 16-bit A 16x32 layout): lane<16 holds K 0..7,16..23 of row l16
      const _Float16* ap = &sA[(k * Bn + l16) * ASTRIDE + h * 32 + (lhi << 3)];
      v8h a_lo = *(const v8h*)(ap);          // ds_load_b128
      v8h a_hi = *(const v8h*)(ap + 16);     // ds_load_b128
      v16h a = __builtin_shufflevector(a_lo, a_hi,
               0,1,2,3,4,5,6,7,8,9,10,11,12,13,14,15);

      // B frag: lane<16 -> column o, K 0..7,16..23 ; lane>=16 -> K 8..15,24..31 (Wt is [k][o][i])
      const _Float16* bp = wt + (((k << 6) + (wv << 4) + l16) << 6) + h * 32 + (lhi << 3);
      v8h b_lo = *(const v8h*)(bp);          // global_load_b128 (L2-hot, 128KB total)
      v8h b_hi = *(const v8h*)(bp + 16);
      v16h b = __builtin_shufflevector(b_lo, b_hi,
               0,1,2,3,4,5,6,7,8,9,10,11,12,13,14,15);

      acc = __builtin_amdgcn_wmma_f32_16x16x32_f16(false, a, false, b,
                                                   (short)0, acc, false, false);
    }
  }

  // C/D layout: VGPR r, lanes 0-15 -> (M=r, N=lane) ; lanes 16-31 -> (M=r+8)
  const int o = (wv << 4) + l16;
#pragma unroll
  for (int r = 0; r < 8; ++r) {
    int m = r + (lhi << 3);                  // batch index
    y[(((size_t)m * Nn + n) << 6) + o] = acc[r];
  }
}

extern "C" void kernel_launch(void* const* d_in, const int* in_sizes, int n_in,
                              void* d_out, int out_size, void* d_ws, size_t ws_size,
                              hipStream_t stream) {
  const float* x   = (const float*)d_in[0];
  const int*   knn = (const int*)d_in[1];
  const float* bs  = (const float*)d_in[2];
  const float* ws  = (const float*)d_in[3];
  float*    y  = (float*)d_out;
  _Float16* wt = (_Float16*)d_ws;   // 128 KB scratch for f16-transposed weights

  prep_wt<<<(Kk * CIN * COUT + 255) / 256, 256, 0, stream>>>(ws, wt);
  slc_kernel<<<Nn, 128, 0, stream>>>(x, knn, bs, wt, y);
}
